// UEDDIEFinetuner_13511967113399
// MI455X (gfx1250) — compile-verified, matrix operand order checked by reading, not run on hardware
//
#include <hip/hip_runtime.h>
#include <hip/hip_bf16.h>

// ---------------- problem constants (match reference) ----------------
#define Bc    32
#define Nc    256
#define Dc    256
#define Hc    4
#define DHc   64          // Dc/Hc
#define DFFc  128
#define Lc    2
#define NEXPc 12
#define ROWS  (Bc * Nc)   // 8192

typedef __attribute__((ext_vector_type(16))) _Float16 v16h;
typedef __attribute__((ext_vector_type(8)))  float    v8f;

// =====================================================================
// WMMA fragment loaders (layouts per CDNA5 ISA §7.12.2)
// A (16x32 f16): lane L holds row M=L&15; half-wave selects K offset 0/8;
//   elements 0..7 -> K = kA+0..7, elements 8..15 -> K = 16+kA+0..7
// B (32x16 f16): mirrored, lane L holds column N=L&15.
// C/D (16x16 f32): VGPR r -> M = r + 8*(L>>4), N = L&15.
// =====================================================================
__device__ __forceinline__ v16h frag_a_f32(const float* __restrict__ A, int lda,
                                           int row, int k0, int lane) {
  int kA = (lane >> 4) << 3;
  const float* p = A + (size_t)row * lda + k0 + kA;
  v16h a;
#pragma unroll
  for (int t = 0; t < 8; ++t) {
    a[t]     = (_Float16)p[t];
    a[8 + t] = (_Float16)p[16 + t];
  }
  return a;
}

__device__ __forceinline__ v16h frag_a_f16(const _Float16* __restrict__ A, int lda,
                                           int row, int k0, int lane) {
  int kA = (lane >> 4) << 3;
  const _Float16* p = A + (size_t)row * lda + k0 + kA;
  v16h a;
#pragma unroll
  for (int t = 0; t < 8; ++t) {
    a[t]     = p[t];
    a[8 + t] = p[16 + t];
  }
  return a;
}

// B fragment from a [n][k] (transposed, K contiguous) f16 matrix
__device__ __forceinline__ v16h frag_bT_f16(const _Float16* __restrict__ BT, int ldb,
                                            int col, int k0, int lane) {
  int kB = (lane >> 4) << 3;
  const _Float16* p = BT + (size_t)col * ldb + k0 + kB;
  v16h b;
#pragma unroll
  for (int t = 0; t < 8; ++t) {
    b[t]     = p[t];
    b[8 + t] = p[16 + t];
  }
  return b;
}

// =====================================================================
// Register-blocked WMMA GEMM: Y[M x Nout] = A(f32) * BT[Nout x K](f16) + bias
// One wave computes a 16 x (16*NB) tile: A fragment loaded once per k-step,
// NB independent accumulators -> NB back-to-back v_wmma per k-step.
// OUTMODE: 0 = f32 row-major, 1 = f16 row-major, 2 = f16 per-batch transposed
//          (vT[b][col][n], n = row within batch; rows-per-batch = Nc)
// =====================================================================
template <int NB, bool RELU, int OUTMODE>
__global__ void wmma_gemm(const float* __restrict__ A, const _Float16* __restrict__ BT,
                          const float* __restrict__ bias, float* __restrict__ Cf,
                          _Float16* __restrict__ Ch, int M, int K, int Nout) {
  int lane = threadIdx.x & 31;
  int wave = threadIdx.x >> 5;
  int tile = blockIdx.x * (blockDim.x >> 5) + wave;
  int ntb = Nout / (16 * NB);
  int mt = tile / ntb;
  int nt = tile - mt * ntb;
  if (mt >= (M >> 4)) return;  // wave-uniform (never taken with exact grids)
  int m0 = mt << 4, n0 = nt * 16 * NB;

  v8f c[NB];
#pragma unroll
  for (int j = 0; j < NB; ++j) c[j] = (v8f){};

  for (int k0 = 0; k0 < K; k0 += 32) {
    v16h a = frag_a_f32(A, K, m0 + (lane & 15), k0, lane);
#pragma unroll
    for (int j = 0; j < NB; ++j) {
      v16h b = frag_bT_f16(BT, K, n0 + j * 16 + (lane & 15), k0, lane);
      c[j] = __builtin_amdgcn_wmma_f32_16x16x32_f16(false, a, false, b, (short)0,
                                                    c[j], false, false);
    }
  }

#pragma unroll
  for (int j = 0; j < NB; ++j) {
    int col = n0 + j * 16 + (lane & 15);
    float bv = bias ? bias[col] : 0.0f;
#pragma unroll
    for (int r = 0; r < 8; ++r) {
      int row = m0 + r + ((lane >> 4) << 3);
      float v = c[j][r] + bv;
      if (RELU) v = v > 0.0f ? v : 0.0f;
      if (OUTMODE == 0) {
        Cf[(size_t)row * Nout + col] = v;
      } else if (OUTMODE == 1) {
        Ch[(size_t)row * Nout + col] = (_Float16)v;
      } else {  // per-batch transposed: vT[(b*Nout + col)*Nc + n]
        int bI = row >> 8;          // Nc == 256 rows per batch
        int n = row & (Nc - 1);
        Ch[((size_t)bI * Nout + col) * Nc + n] = (_Float16)v;
      }
    }
  }
}

// =====================================================================
// Fused attention: one wave per (b, h, 16-row q tile).
// scores (WMMA, hoisted Q frags) -> LDS softmax -> f16 probs ->
// P @ V via transposed V (contiguous B frags), hoisted P frags.
// =====================================================================
__global__ void __launch_bounds__(32)
attn_kernel(const _Float16* __restrict__ q, const _Float16* __restrict__ k,
            const _Float16* __restrict__ vT, float* __restrict__ ctx) {
  __shared__ float    sS[16][Nc];
  __shared__ _Float16 sP[16][Nc];

  int lane = threadIdx.x;
  int id = blockIdx.x;
  int qt = id & ((Nc / 16) - 1);
  int h  = (id / (Nc / 16)) & (Hc - 1);
  int b  = id / ((Nc / 16) * Hc);
  int q0 = qt << 4;

  const _Float16* qb  = q  + ((size_t)b * Nc) * Dc + h * DHc;          // row stride Dc
  const _Float16* kb  = k  + ((size_t)b * Nc) * Dc + h * DHc;
  const _Float16* vTb = vT + ((size_t)b * Dc + h * DHc) * Nc;          // [d][n], n contiguous

  // ---- hoisted Q fragments (loop-invariant across score column tiles) ----
  v16h aq[DHc / 32];
#pragma unroll
  for (int ks = 0; ks < DHc / 32; ++ks)
    aq[ks] = frag_a_f16(qb, Dc, q0 + (lane & 15), ks * 32, lane);

  // ---- scores: S = (Q K^T) / sqrt(dh) ----
  for (int nt = 0; nt < Nc / 16; ++nt) {
    v8f c = {};
#pragma unroll
    for (int ks = 0; ks < DHc / 32; ++ks) {
      v16h bfr = frag_bT_f16(kb, Dc, (nt << 4) + (lane & 15), ks * 32, lane);
      c = __builtin_amdgcn_wmma_f32_16x16x32_f16(false, aq[ks], false, bfr, (short)0,
                                                 c, false, false);
    }
    int col = (nt << 4) + (lane & 15);
#pragma unroll
    for (int r = 0; r < 8; ++r)
      sS[r + ((lane >> 4) << 3)][col] = c[r] * 0.125f;  // 1/sqrt(64)
  }
  __syncthreads();

  // ---- softmax along rows of 256 (32 lanes x 8 each) ----
  for (int r = 0; r < 16; ++r) {
    float m = -3.0e38f;
    for (int cI = lane; cI < Nc; cI += 32) m = fmaxf(m, sS[r][cI]);
#pragma unroll
    for (int off = 16; off > 0; off >>= 1) m = fmaxf(m, __shfl_xor(m, off, 32));
    float s = 0.0f;
    for (int cI = lane; cI < Nc; cI += 32) {
      float e = __expf(sS[r][cI] - m);
      sS[r][cI] = e;
      s += e;
    }
#pragma unroll
    for (int off = 16; off > 0; off >>= 1) s += __shfl_xor(s, off, 32);
    float inv = 1.0f / s;
    for (int cI = lane; cI < Nc; cI += 32) sP[r][cI] = (_Float16)(sS[r][cI] * inv);
  }
  __syncthreads();

  // ---- hoisted P fragments (loop-invariant across output tiles) ----
  v16h ap[Nc / 32];
#pragma unroll
  for (int ks = 0; ks < Nc / 32; ++ks)
    ap[ks] = frag_a_f16(&sP[0][0], Nc, lane & 15, ks * 32, lane);

  // ---- context = P @ V  (V^T gives contiguous B fragments) ----
  for (int dt = 0; dt < DHc / 16; ++dt) {
    v8f c = {};
#pragma unroll
    for (int ks = 0; ks < Nc / 32; ++ks) {
      v16h bfr = frag_bT_f16(vTb, Nc, (dt << 4) + (lane & 15), ks * 32, lane);
      c = __builtin_amdgcn_wmma_f32_16x16x32_f16(false, ap[ks], false, bfr, (short)0,
                                                 c, false, false);
    }
    int col = h * DHc + (dt << 4) + (lane & 15);
#pragma unroll
    for (int r = 0; r < 8; ++r) {
      int row = q0 + r + ((lane >> 4) << 3);
      ctx[((size_t)b * Nc + row) * Dc + col] = c[r];
    }
  }
}

// =====================================================================
// x = LayerNorm(x + y) * g + be      (one wave per 256-wide row)
// =====================================================================
__global__ void res_ln_kernel(float* __restrict__ x, const float* __restrict__ y,
                              const float* __restrict__ g, const float* __restrict__ be) {
  int lane = threadIdx.x & 31;
  int row = blockIdx.x * (blockDim.x >> 5) + (threadIdx.x >> 5);
  const size_t base = (size_t)row * Dc;
  float s1 = 0.0f, s2 = 0.0f;
  for (int cI = lane; cI < Dc; cI += 32) {
    float xv = x[base + cI] + y[base + cI];
    s1 += xv;
    s2 += xv * xv;
  }
#pragma unroll
  for (int off = 16; off > 0; off >>= 1) {
    s1 += __shfl_xor(s1, off, 32);
    s2 += __shfl_xor(s2, off, 32);
  }
  float m = s1 * (1.0f / Dc);
  float var = s2 * (1.0f / Dc) - m * m;
  float inv = rsqrtf(var + 1e-5f);
  for (int cI = lane; cI < Dc; cI += 32) {
    float xv = x[base + cI] + y[base + cI];
    x[base + cI] = (xv - m) * inv * g[cI] + be[cI];
  }
}

// =====================================================================
// masked init:  x = (bucket == e) ? X : 0
// =====================================================================
__global__ void mask_init_kernel(const float* __restrict__ X, const int* __restrict__ E,
                                 const int* __restrict__ C, float* __restrict__ x, int e) {
  int idx = blockIdx.x * blockDim.x + threadIdx.x;
  if (idx >= ROWS * Dc) return;
  int atom = idx / Dc;
  int bucket = E[atom] * 3 + (C[atom] + 1);
  x[idx] = (bucket == e) ? X[idx] : 0.0f;
}

// =====================================================================
// head:  acc[atom] = x_row . Wh[e] + bh[e]   (only where bucket == e)
// =====================================================================
__global__ void head_kernel(const float* __restrict__ x, const float* __restrict__ Wh,
                            const float* __restrict__ bh, const int* __restrict__ E,
                            const int* __restrict__ C, float* __restrict__ acc, int e) {
  int lane = threadIdx.x & 31;
  int atom = blockIdx.x * (blockDim.x >> 5) + (threadIdx.x >> 5);
  const float* w = Wh + (size_t)e * Dc;
  float s = 0.0f;
  for (int cI = lane; cI < Dc; cI += 32)
    s += x[(size_t)atom * Dc + cI] * w[cI];
#pragma unroll
  for (int off = 16; off > 0; off >>= 1) s += __shfl_xor(s, off, 32);
  int bucket = E[atom] * 3 + (C[atom] + 1);
  if (lane == 0 && bucket == e) acc[atom] = s + bh[e];
}

// =====================================================================
// out[b] = -sum_n acc[b][n]
// =====================================================================
__global__ void final_kernel(const float* __restrict__ acc, float* __restrict__ out) {
  __shared__ float red[8];
  int b = blockIdx.x;
  int tid = threadIdx.x;
  float s = acc[(size_t)b * Nc + tid];
#pragma unroll
  for (int off = 16; off > 0; off >>= 1) s += __shfl_xor(s, off, 32);
  if ((tid & 31) == 0) red[tid >> 5] = s;
  __syncthreads();
  if (tid == 0) {
    float t = 0.0f;
#pragma unroll
    for (int w = 0; w < 8; ++w) t += red[w];
    out[b] = -t;
  }
}

// =====================================================================
// convert W (count x K x Nout, f32, row-major) -> WT (count x Nout x K, f16)
// =====================================================================
__global__ void convert_w_kernel(const float* __restrict__ W, _Float16* __restrict__ WT,
                                 int K, int Nout, int total) {
  int idx = blockIdx.x * blockDim.x + threadIdx.x;
  if (idx >= total) return;
  int t = idx / (K * Nout);
  int rem = idx - t * (K * Nout);
  int kk = rem / Nout;
  int n = rem - kk * Nout;
  WT[((size_t)t * Nout + n) * K + kk] = (_Float16)W[idx];
}

// =====================================================================
// host launcher
// =====================================================================
extern "C" void kernel_launch(void* const* d_in, const int* in_sizes, int n_in,
                              void* d_out, int out_size, void* d_ws, size_t ws_size,
                              hipStream_t stream) {
  const float* X   = (const float*)d_in[0];
  const int*   E   = (const int*)d_in[1];
  const int*   C   = (const int*)d_in[2];
  const float* Wq  = (const float*)d_in[3];
  const float* bq  = (const float*)d_in[4];
  const float* Wk  = (const float*)d_in[5];
  const float* bk  = (const float*)d_in[6];
  const float* Wv  = (const float*)d_in[7];
  const float* bv  = (const float*)d_in[8];
  const float* Wo  = (const float*)d_in[9];
  const float* bo  = (const float*)d_in[10];
  const float* g1  = (const float*)d_in[11];
  const float* be1 = (const float*)d_in[12];
  const float* W1  = (const float*)d_in[13];
  const float* b1  = (const float*)d_in[14];
  const float* W2  = (const float*)d_in[15];
  const float* b2  = (const float*)d_in[16];
  const float* g2  = (const float*)d_in[17];
  const float* be2 = (const float*)d_in[18];
  const float* Wh  = (const float*)d_in[19];
  const float* bh  = (const float*)d_in[20];
  float* out = (float*)d_out;

  // ---- bump-allocate scratch (~56 MB total) ----
  char* p = (char*)d_ws;
  auto alloc = [&](size_t bytes) {
    void* r = (void*)p;
    p += (bytes + 255) & ~(size_t)255;
    return r;
  };
  const size_t nWL = (size_t)NEXPc * Lc;
  _Float16* WqT = (_Float16*)alloc(nWL * Dc * Dc * 2);
  _Float16* WkT = (_Float16*)alloc(nWL * Dc * Dc * 2);
  _Float16* WvT = (_Float16*)alloc(nWL * Dc * Dc * 2);
  _Float16* WoT = (_Float16*)alloc(nWL * Dc * Dc * 2);
  _Float16* W1T = (_Float16*)alloc(nWL * Dc * DFFc * 2);
  _Float16* W2T = (_Float16*)alloc(nWL * DFFc * Dc * 2);
  float*    xb  = (float*)alloc((size_t)ROWS * Dc * 4);
  _Float16* qb  = (_Float16*)alloc((size_t)ROWS * Dc * 2);
  _Float16* kb  = (_Float16*)alloc((size_t)ROWS * Dc * 2);
  _Float16* vTb = (_Float16*)alloc((size_t)ROWS * Dc * 2);  // [b][d][n]
  float*    ctx = (float*)alloc((size_t)ROWS * Dc * 4);
  float*    yb  = (float*)alloc((size_t)ROWS * Dc * 4);
  float*    hb  = (float*)alloc((size_t)ROWS * DFFc * 4);
  float*    acc = (float*)alloc((size_t)ROWS * 4);

  // ---- weight convert / transpose (f32 -> f16 [n][k]) ----
  {
    int totDD = (int)(nWL * Dc * Dc);
    int totDF = (int)(nWL * Dc * DFFc);
    int gDD = (totDD + 255) / 256, gDF = (totDF + 255) / 256;
    convert_w_kernel<<<gDD, 256, 0, stream>>>(Wq, WqT, Dc, Dc, totDD);
    convert_w_kernel<<<gDD, 256, 0, stream>>>(Wk, WkT, Dc, Dc, totDD);
    convert_w_kernel<<<gDD, 256, 0, stream>>>(Wv, WvT, Dc, Dc, totDD);
    convert_w_kernel<<<gDD, 256, 0, stream>>>(Wo, WoT, Dc, Dc, totDD);
    convert_w_kernel<<<gDF, 256, 0, stream>>>(W1, W1T, Dc, DFFc, totDF);
    convert_w_kernel<<<gDF, 256, 0, stream>>>(W2, W2T, DFFc, Dc, totDF);
  }

  // waves = (M/16) * (Nout/64); 8 waves per 256-thread block
  const int gemmDD_blocks = (ROWS / 16) * (Dc / 64) / 8;    // 256
  const int gemmDF_blocks = (ROWS / 16) * (DFFc / 64) / 8;  // 128
  const int attn_blocks = Bc * Hc * (Nc / 16);              // 2048
  const int row_blocks = ROWS / 8;                          // 1024

  for (int e = 0; e < NEXPc; ++e) {
    mask_init_kernel<<<(ROWS * Dc + 255) / 256, 256, 0, stream>>>(X, E, C, xb, e);

    for (int l = 0; l < Lc; ++l) {
      size_t w = (size_t)e * Lc + l;
      // QKV projections (bias fused; V written per-batch transposed)
      wmma_gemm<4, false, 1><<<gemmDD_blocks, 256, 0, stream>>>(
          xb, WqT + w * Dc * Dc, bq + w * Dc, nullptr, qb, ROWS, Dc, Dc);
      wmma_gemm<4, false, 1><<<gemmDD_blocks, 256, 0, stream>>>(
          xb, WkT + w * Dc * Dc, bk + w * Dc, nullptr, kb, ROWS, Dc, Dc);
      wmma_gemm<4, false, 2><<<gemmDD_blocks, 256, 0, stream>>>(
          xb, WvT + w * Dc * Dc, bv + w * Dc, nullptr, vTb, ROWS, Dc, Dc);
      // attention
      attn_kernel<<<attn_blocks, 32, 0, stream>>>(qb, kb, vTb, ctx);
      // output projection
      wmma_gemm<4, false, 0><<<gemmDD_blocks, 256, 0, stream>>>(
          ctx, WoT + w * Dc * Dc, bo + w * Dc, yb, nullptr, ROWS, Dc, Dc);
      // x = LN(x + attn_out)
      res_ln_kernel<<<row_blocks, 256, 0, stream>>>(xb, yb, g1 + w * Dc, be1 + w * Dc);
      // FFN
      wmma_gemm<4, true, 0><<<gemmDF_blocks, 256, 0, stream>>>(
          xb, W1T + w * Dc * DFFc, b1 + w * DFFc, hb, nullptr, ROWS, Dc, DFFc);
      wmma_gemm<4, false, 0><<<gemmDD_blocks, 256, 0, stream>>>(
          hb, W2T + w * DFFc * Dc, b2 + w * Dc, yb, nullptr, ROWS, DFFc, Dc);
      res_ln_kernel<<<row_blocks, 256, 0, stream>>>(xb, yb, g2 + w * Dc, be2 + w * Dc);
    }
    head_kernel<<<row_blocks, 256, 0, stream>>>(xb, Wh, bh, E, C, acc, e);
  }

  final_kernel<<<Bc, 256, 0, stream>>>(acc, out);
}
